// NNDistance_34050500722812
// MI455X (gfx1250) — matrix operand order repaired and checked by reading, not compiled, not run on hardware
//
#include <hip/hip_runtime.h>

#define NROWS 4096
#define MCOLS 65536
#define DDIM  256
#define MCHUNKS 16
#define MPERCHUNK (MCOLS / MCHUNKS)   // 4096
#define NTILES (MPERCHUNK / 16)       // 256
#define ROWBLK 128
#define NBLOCKS (NROWS / ROWBLK)      // 32
#define KPAD 264                      // 256 + 8 bf16 pad (bank-conflict stagger)

typedef __attribute__((ext_vector_type(16))) __bf16          v16bf;
typedef __attribute__((ext_vector_type(16))) unsigned short  v16us;
typedef __attribute__((ext_vector_type(8)))  unsigned short  v8us;
typedef __attribute__((ext_vector_type(8)))  float           v8f;
typedef __attribute__((ext_vector_type(4)))  float           v4f;

__device__ __forceinline__ unsigned short f32_bf16_rn(float f) {
  unsigned u = __float_as_uint(f);
  unsigned r = u + 0x7FFFu + ((u >> 16) & 1u);
  return (unsigned short)(r >> 16);
}
__device__ __forceinline__ float bf16_f32(unsigned short h) {
  return __uint_as_float(((unsigned)h) << 16);
}

// async copy of 16B global -> LDS (no VGPR data path, ASYNCcnt-tracked)
__device__ __forceinline__ void async_cp16(unsigned lds_off, const void* sbase, unsigned voff) {
  asm volatile("global_load_async_to_lds_b128 %0, %1, %2"
               :: "v"(lds_off), "v"(voff), "s"(sbase) : "memory");
}
__device__ __forceinline__ void wait_asynccnt0() {
  asm volatile("s_wait_asynccnt 0" ::: "memory");
}

// -------- kernel 1: row norms for target (M rows) and x (N rows); wave per row
__global__ void nn_norms(const float* __restrict__ x, const float* __restrict__ tgt,
                         float* __restrict__ tn, float* __restrict__ xn) {
  const int wave = threadIdx.x >> 5;
  const int lane = threadIdx.x & 31;
  const int row  = blockIdx.x * 8 + wave;
  const float* p;
  float* o;
  if (row < MCOLS) { p = tgt + (size_t)row * DDIM; o = tn + row; }
  else             { int r = row - MCOLS; p = x + (size_t)r * DDIM; o = xn + r; }
  float s = 0.f;
#pragma unroll
  for (int c = 0; c < 2; ++c) {
    v4f v = *(const v4f*)(p + c * 128 + lane * 4);
    s += v.x * v.x + v.y * v.y + v.z * v.z + v.w * v.w;
  }
#pragma unroll
  for (int off = 16; off >= 1; off >>= 1) s += __shfl_xor(s, off, 32);
  if (lane == 0) *o = s;
}

// -------- kernel 1b: one-time fp32 -> bf16 hi/lo plane split of the target set
__global__ void nn_convert(const float* __restrict__ tgt,
                           unsigned short* __restrict__ hiP, unsigned short* __restrict__ loP) {
  const size_t base = ((size_t)blockIdx.x * 256 + threadIdx.x) * 16;
  float f[16];
#pragma unroll
  for (int q = 0; q < 4; ++q) *(v4f*)&f[q * 4] = *(const v4f*)(tgt + base + q * 4);
  unsigned ho[8], lo[8];
#pragma unroll
  for (int j = 0; j < 8; ++j) {
    unsigned short h0 = f32_bf16_rn(f[2 * j]);
    unsigned short h1 = f32_bf16_rn(f[2 * j + 1]);
    unsigned short l0 = f32_bf16_rn(f[2 * j] - bf16_f32(h0));
    unsigned short l1 = f32_bf16_rn(f[2 * j + 1] - bf16_f32(h1));
    ho[j] = (unsigned)h0 | ((unsigned)h1 << 16);
    lo[j] = (unsigned)l0 | ((unsigned)l1 << 16);
  }
#pragma unroll
  for (int j = 0; j < 8; ++j) {
    ((unsigned*)(hiP + base))[j] = ho[j];
    ((unsigned*)(loP + base))[j] = lo[j];
  }
}

// ======== kernel 2 (fast path): WMMA GEMM + fused min/argmin,
//          B tiles streamed global(bf16 planes) -> LDS with async-to-LDS copies
__global__ void __launch_bounds__(256)
nn_main_pre(const float* __restrict__ x,
            const unsigned short* __restrict__ hiP, const unsigned short* __restrict__ loP,
            const float* __restrict__ tn, float* __restrict__ pval, int* __restrict__ pidx) {
  __shared__ unsigned short sB[2][2][16][KPAD];   // [plane hi/lo][buf][col][k]
  __shared__ float sTn[2][16];

  const int tid    = threadIdx.x;
  const int wave   = tid >> 5;
  const int lane   = tid & 31;
  const int nblock = blockIdx.x;
  const int mchunk = blockIdx.y;
  const int mbase  = mchunk * MPERCHUNK;

  // ---- A fragments: this wave's 16 x-rows, fp32 -> bf16 hi/lo, VGPR-resident
  const int rloc = wave * 16 + (lane & 15);
  const int rowG = nblock * ROWBLK + rloc;
  const int koff = (lane >> 4) * 16;
  const float* xr = x + (size_t)rowG * DDIM;

  v16us a_hi[8], a_lo[8];
#pragma unroll
  for (int ks = 0; ks < 8; ++ks) {
    float f[16];
    *(v4f*)&f[0]  = *(const v4f*)(xr + ks * 32 + koff);
    *(v4f*)&f[4]  = *(const v4f*)(xr + ks * 32 + koff + 4);
    *(v4f*)&f[8]  = *(const v4f*)(xr + ks * 32 + koff + 8);
    *(v4f*)&f[12] = *(const v4f*)(xr + ks * 32 + koff + 12);
#pragma unroll
    for (int j = 0; j < 16; ++j) {
      unsigned short h = f32_bf16_rn(f[j]);
      a_hi[ks][j] = h;
      a_lo[ks][j] = f32_bf16_rn(f[j] - bf16_f32(h));
    }
  }

  float best[8];
  int   bidx[8];
#pragma unroll
  for (int r = 0; r < 8; ++r) { best[r] = 3.4e38f; bidx[r] = 0; }

  // copy assignment: plane = tid>>7 (wave-uniform), row = (tid>>3)&15, seg = tid&7 (64B)
  const int cpl  = tid >> 7;
  const int crow = (tid >> 3) & 15;
  const int cseg = tid & 7;
  const unsigned short* plane = cpl ? loP : hiP;

  auto issueCopy = [&](int tile, int buf) {
    const unsigned gvoff =
        (unsigned)(((size_t)(mbase + tile * 16 + crow) * DDIM + cseg * 32) * sizeof(unsigned short));
    const unsigned lbase = (unsigned)(size_t)&sB[cpl][buf][crow][cseg * 32];
#pragma unroll
    for (int j = 0; j < 4; ++j) async_cp16(lbase + j * 16, plane, gvoff + j * 16);
    if (tid < 16) sTn[buf][tid] = tn[mbase + tile * 16 + tid];
  };

  issueCopy(0, 0);
  wait_asynccnt0();
  __syncthreads();

  const int n  = lane & 15;
  const int kb = (lane >> 4) * 16;   // B layout: lanes 0-15 K 0..15, lanes 16-31 K 16..31

  for (int tile = 0; tile < NTILES; ++tile) {
    const int cur = tile & 1;

    if (tile + 1 < NTILES) issueCopy(tile + 1, cur ^ 1);  // async, overlaps WMMAs

    v8f acc = {};
#pragma unroll
    for (int ks = 0; ks < 8; ++ks) {
      const v8us* bh = (const v8us*)&sB[0][cur][n][ks * 32 + kb];
      const v8us* bl = (const v8us*)&sB[1][cur][n][ks * 32 + kb];
      v16us bhi = __builtin_shufflevector(bh[0], bh[1], 0,1,2,3,4,5,6,7,8,9,10,11,12,13,14,15);
      v16us blo = __builtin_shufflevector(bl[0], bl[1], 0,1,2,3,4,5,6,7,8,9,10,11,12,13,14,15);
      acc = __builtin_amdgcn_wmma_f32_16x16x32_bf16(
          false, __builtin_bit_cast(v16bf, a_hi[ks]),
          false, __builtin_bit_cast(v16bf, bhi), (short)0, acc, false, false);
      acc = __builtin_amdgcn_wmma_f32_16x16x32_bf16(
          false, __builtin_bit_cast(v16bf, a_hi[ks]),
          false, __builtin_bit_cast(v16bf, blo), (short)0, acc, false, false);
      acc = __builtin_amdgcn_wmma_f32_16x16x32_bf16(
          false, __builtin_bit_cast(v16bf, a_lo[ks]),
          false, __builtin_bit_cast(v16bf, bhi), (short)0, acc, false, false);
    }

    const float tnv = sTn[cur][lane & 15];
    const int   col = mbase + tile * 16 + (lane & 15);
#pragma unroll
    for (int r = 0; r < 8; ++r) {
      float s = fmaf(-2.f, acc[r], tnv);
      if (s < best[r]) { best[r] = s; bidx[r] = col; }
    }

    wait_asynccnt0();    // next buffer's async copies landed
    __syncthreads();
  }

#pragma unroll
  for (int r = 0; r < 8; ++r) {
#pragma unroll
    for (int off = 8; off >= 1; off >>= 1) {
      float ov = __shfl_xor(best[r], off, 16);
      int   oi = __shfl_xor(bidx[r], off, 16);
      if (ov < best[r] || (ov == best[r] && oi < bidx[r])) { best[r] = ov; bidx[r] = oi; }
    }
  }
  if ((lane & 15) == 0) {
    const int half = lane >> 4;
#pragma unroll
    for (int r = 0; r < 8; ++r) {
      int row = nblock * ROWBLK + wave * 16 + r + half * 8;
      pval[(size_t)mchunk * NROWS + row] = best[r];
      pidx[(size_t)mchunk * NROWS + row] = bidx[r];
    }
  }
}

// ======== kernel 2 (fallback, small ws): in-kernel conversion, double-buffered
__global__ void __launch_bounds__(256)
nn_main_conv(const float* __restrict__ x, const float* __restrict__ tgt,
             const float* __restrict__ tn, float* __restrict__ pval, int* __restrict__ pidx) {
  __shared__ unsigned short sBhi[2][16][KPAD];
  __shared__ unsigned short sBlo[2][16][KPAD];
  __shared__ float sTn[2][16];

  const int tid    = threadIdx.x;
  const int wave   = tid >> 5;
  const int lane   = tid & 31;
  const int nblock = blockIdx.x;
  const int mchunk = blockIdx.y;
  const int mbase  = mchunk * MPERCHUNK;

  const int rloc = wave * 16 + (lane & 15);
  const int rowG = nblock * ROWBLK + rloc;
  const int koff = (lane >> 4) * 16;
  const float* xr = x + (size_t)rowG * DDIM;

  v16us a_hi[8], a_lo[8];
#pragma unroll
  for (int ks = 0; ks < 8; ++ks) {
    float f[16];
    *(v4f*)&f[0]  = *(const v4f*)(xr + ks * 32 + koff);
    *(v4f*)&f[4]  = *(const v4f*)(xr + ks * 32 + koff + 4);
    *(v4f*)&f[8]  = *(const v4f*)(xr + ks * 32 + koff + 8);
    *(v4f*)&f[12] = *(const v4f*)(xr + ks * 32 + koff + 12);
#pragma unroll
    for (int j = 0; j < 16; ++j) {
      unsigned short h = f32_bf16_rn(f[j]);
      a_hi[ks][j] = h;
      a_lo[ks][j] = f32_bf16_rn(f[j] - bf16_f32(h));
    }
  }

  float best[8];
  int   bidx[8];
#pragma unroll
  for (int r = 0; r < 8; ++r) { best[r] = 3.4e38f; bidx[r] = 0; }

  const int ccol = tid >> 4;
  const int ckc  = (tid & 15) * 16;

  auto convertTile = [&](int tile, int buf) {
    const float* tp = tgt + ((size_t)(mbase + tile * 16 + ccol)) * DDIM + ckc;
    float f[16];
#pragma unroll
    for (int q = 0; q < 4; ++q) *(v4f*)&f[q * 4] = *(const v4f*)(tp + q * 4);
    unsigned* dhi = (unsigned*)&sBhi[buf][ccol][ckc];
    unsigned* dlo = (unsigned*)&sBlo[buf][ccol][ckc];
#pragma unroll
    for (int j = 0; j < 8; ++j) {
      unsigned short h0 = f32_bf16_rn(f[2 * j]);
      unsigned short h1 = f32_bf16_rn(f[2 * j + 1]);
      unsigned short l0 = f32_bf16_rn(f[2 * j] - bf16_f32(h0));
      unsigned short l1 = f32_bf16_rn(f[2 * j + 1] - bf16_f32(h1));
      dhi[j] = (unsigned)h0 | ((unsigned)h1 << 16);
      dlo[j] = (unsigned)l0 | ((unsigned)l1 << 16);
    }
    if (tid < 16) sTn[buf][tid] = tn[mbase + tile * 16 + tid];
  };

  convertTile(0, 0);
  __syncthreads();

  const int n  = lane & 15;
  const int kb = (lane >> 4) * 16;

  for (int tile = 0; tile < NTILES; ++tile) {
    const int cur = tile & 1;
    if (tile + 1 < NTILES) convertTile(tile + 1, cur ^ 1);

    v8f acc = {};
#pragma unroll
    for (int ks = 0; ks < 8; ++ks) {
      const v8us* bh = (const v8us*)&sBhi[cur][n][ks * 32 + kb];
      const v8us* bl = (const v8us*)&sBlo[cur][n][ks * 32 + kb];
      v16us bhi = __builtin_shufflevector(bh[0], bh[1], 0,1,2,3,4,5,6,7,8,9,10,11,12,13,14,15);
      v16us blo = __builtin_shufflevector(bl[0], bl[1], 0,1,2,3,4,5,6,7,8,9,10,11,12,13,14,15);
      acc = __builtin_amdgcn_wmma_f32_16x16x32_bf16(
          false, __builtin_bit_cast(v16bf, a_hi[ks]),
          false, __builtin_bit_cast(v16bf, bhi), (short)0, acc, false, false);
      acc = __builtin_amdgcn_wmma_f32_16x16x32_bf16(
          false, __builtin_bit_cast(v16bf, a_hi[ks]),
          false, __builtin_bit_cast(v16bf, blo), (short)0, acc, false, false);
      acc = __builtin_amdgcn_wmma_f32_16x16x32_bf16(
          false, __builtin_bit_cast(v16bf, a_lo[ks]),
          false, __builtin_bit_cast(v16bf, bhi), (short)0, acc, false, false);
    }

    const float tnv = sTn[cur][lane & 15];
    const int   col = mbase + tile * 16 + (lane & 15);
#pragma unroll
    for (int r = 0; r < 8; ++r) {
      float s = fmaf(-2.f, acc[r], tnv);
      if (s < best[r]) { best[r] = s; bidx[r] = col; }
    }
    __syncthreads();
  }

#pragma unroll
  for (int r = 0; r < 8; ++r) {
#pragma unroll
    for (int off = 8; off >= 1; off >>= 1) {
      float ov = __shfl_xor(best[r], off, 16);
      int   oi = __shfl_xor(bidx[r], off, 16);
      if (ov < best[r] || (ov == best[r] && oi < bidx[r])) { best[r] = ov; bidx[r] = oi; }
    }
  }
  if ((lane & 15) == 0) {
    const int half = lane >> 4;
#pragma unroll
    for (int r = 0; r < 8; ++r) {
      int row = nblock * ROWBLK + wave * 16 + r + half * 8;
      pval[(size_t)mchunk * NROWS + row] = best[r];
      pidx[(size_t)mchunk * NROWS + row] = bidx[r];
    }
  }
}

// -------- kernel 3: reduce 16 chunk partials per row, add ||x||^2, sqrt
__global__ void nn_reduce(const float* __restrict__ pval, const int* __restrict__ pidx,
                          const float* __restrict__ xn,
                          float* __restrict__ outv, float* __restrict__ outi) {
  const int row = blockIdx.x * blockDim.x + threadIdx.x;
  if (row >= NROWS) return;
  float best = 3.4e38f;
  int   bi   = 0;
#pragma unroll
  for (int c = 0; c < MCHUNKS; ++c) {
    float v = pval[(size_t)c * NROWS + row];
    int   i = pidx[(size_t)c * NROWS + row];
    if (v < best || (v == best && i < bi)) { best = v; bi = i; }
  }
  float d2 = xn[row] + best;
  outv[row] = sqrtf(fmaxf(d2, 0.f));
  outi[row] = (float)bi;
}

extern "C" void kernel_launch(void* const* d_in, const int* in_sizes, int n_in,
                              void* d_out, int out_size, void* d_ws, size_t ws_size,
                              hipStream_t stream) {
  const float* x   = (const float*)d_in[0];
  const float* tgt = (const float*)d_in[1];

  float* ws   = (float*)d_ws;
  float* tn   = ws;                                   // [M]
  float* xn   = tn + MCOLS;                           // [N]
  float* pval = xn + NROWS;                           // [MCHUNKS][N]
  int*   pidx = (int*)(pval + (size_t)MCHUNKS * NROWS);
  unsigned short* hiP = (unsigned short*)(pidx + (size_t)MCHUNKS * NROWS);  // [M][D]
  unsigned short* loP = hiP + (size_t)MCOLS * DDIM;                          // [M][D]

  const size_t wsNeeded = ((size_t)(MCOLS + NROWS + 2 * MCHUNKS * NROWS)) * 4
                        + 2 * (size_t)MCOLS * DDIM * 2;

  float* outv = (float*)d_out;
  float* outi = outv + NROWS;

  hipLaunchKernelGGL(nn_norms, dim3((MCOLS + NROWS) / 8), dim3(256), 0, stream, x, tgt, tn, xn);
  if (ws_size >= wsNeeded) {
    hipLaunchKernelGGL(nn_convert, dim3((MCOLS * DDIM) / (256 * 16)), dim3(256), 0, stream,
                       tgt, hiP, loP);
    hipLaunchKernelGGL(nn_main_pre, dim3(NBLOCKS, MCHUNKS), dim3(256), 0, stream,
                       x, hiP, loP, tn, pval, pidx);
  } else {
    hipLaunchKernelGGL(nn_main_conv, dim3(NBLOCKS, MCHUNKS), dim3(256), 0, stream,
                       x, tgt, tn, pval, pidx);
  }
  hipLaunchKernelGGL(nn_reduce, dim3(NROWS / 256), dim3(256), 0, stream, pval, pidx, xn, outv, outi);
}
